// SparseMOE_67869073211797
// MI455X (gfx1250) — matrix operand order repaired and checked by reading, not compiled, not run on hardware
//
#include <hip/hip_runtime.h>
#include <hip/hip_bf16.h>

#define E_EXPERTS 8
#define D_DIM 1024
#define TILE_M 128
#define TILE_N 128
#define TILE_K 32
#define RSTRIDE 17            // padded row stride in dwords (16 data + 1 pad)

typedef __attribute__((ext_vector_type(16))) __bf16        v16bf;
typedef __attribute__((ext_vector_type(8)))  float         v8f;
typedef __attribute__((ext_vector_type(8)))  unsigned int  v8u;
typedef __attribute__((ext_vector_type(4)))  unsigned int  u32x4;
typedef __attribute__((ext_vector_type(8)))  int           i32x8;
typedef __attribute__((ext_vector_type(4)))  int           i32x4;

union FragU { v8u u; v16bf h; };

__device__ __forceinline__ unsigned short f2bf(float f) {
  unsigned int u = __builtin_bit_cast(unsigned int, f);
  u += 0x7fffu + ((u >> 16) & 1u);   // RNE
  return (unsigned short)(u >> 16);
}
__device__ __forceinline__ unsigned int pack2(float lo, float hi) {
  return (unsigned int)f2bf(lo) | ((unsigned int)f2bf(hi) << 16);
}
__device__ __forceinline__ unsigned int lds_off(const void* p) {
  return (unsigned int)(size_t)p;    // generic LDS addr: low 32 bits = offset
}

// ---------------------------------------------------------------------------
// TDM issue: D# built per CDNA5 ISA.
// data_size=2B, pad every 16 dwords by 1 dword -> 17-dword LDS rows.
// gather g2/g3 carry 16x 16-bit row indices when g0_flags has gather bit.
// 6-arg builtin form: (g0 u32x4, g1 i32x8, g2 i32x4, g3 i32x4, i32x8, cpol)
// ---------------------------------------------------------------------------
__device__ __forceinline__ void tdm_issue(unsigned g0_flags, unsigned ldsaddr,
                                          unsigned long long ga,
                                          unsigned tile_d0, unsigned tile_d1,
                                          unsigned stride0, const unsigned* idx) {
  const unsigned DIMBIG = 1u << 30;  // huge tensor dims: no OOB clipping
  u32x4 g0;
  g0.x = (unsigned)__builtin_amdgcn_readfirstlane((int)(g0_flags | 1u)); // count=1
  g0.y = (unsigned)__builtin_amdgcn_readfirstlane((int)ldsaddr);
  g0.z = (unsigned)__builtin_amdgcn_readfirstlane((int)(unsigned)ga);
  g0.w = (unsigned)__builtin_amdgcn_readfirstlane(
      (int)((((unsigned)(ga >> 32)) & 0x01ffffffu) | 0x80000000u)); // type=2
  i32x8 g1;
  g1[0] = (int)((1u << 16) | (1u << 20) | (3u << 22)); // 2B elems, pad_en, every 16 dw, +1 dw
  g1[1] = (int)((DIMBIG & 0xffffu) << 16);             // tensor_dim0 lo
  g1[2] = (int)((DIMBIG >> 16) | ((DIMBIG & 0xffffu) << 16));
  g1[3] = __builtin_amdgcn_readfirstlane((int)((DIMBIG >> 16) | (tile_d0 << 16)));
  g1[4] = __builtin_amdgcn_readfirstlane((int)tile_d1); // tile_dim1 | tile_dim2=0
  g1[5] = __builtin_amdgcn_readfirstlane((int)stride0); // tensor_dim0_stride lo32
  g1[6] = 0;
  g1[7] = 0;
  i32x4 g2, g3;
  if (idx) {
    g2[0] = __builtin_amdgcn_readfirstlane((int)idx[0]);
    g2[1] = __builtin_amdgcn_readfirstlane((int)idx[1]);
    g2[2] = __builtin_amdgcn_readfirstlane((int)idx[2]);
    g2[3] = __builtin_amdgcn_readfirstlane((int)idx[3]);
    g3[0] = __builtin_amdgcn_readfirstlane((int)idx[4]);
    g3[1] = __builtin_amdgcn_readfirstlane((int)idx[5]);
    g3[2] = __builtin_amdgcn_readfirstlane((int)idx[6]);
    g3[3] = __builtin_amdgcn_readfirstlane((int)idx[7]);
  } else {
    g2 = (i32x4)0; g3 = (i32x4)0;
  }
  i32x8 g4 = (i32x8)0;
  __builtin_amdgcn_tensor_load_to_lds(g0, g1, g2, g3, g4, 0);
}

// ---------------------------------------------------------------------------
// Kernel 1: router (one wave32 per token) + fused x -> bf16 conversion.
// ---------------------------------------------------------------------------
__global__ __launch_bounds__(256) void moe_router(
    const float* __restrict__ x, const float* __restrict__ gate_w,
    const float* __restrict__ gate_b, float* __restrict__ logits_out,
    int* __restrict__ counts, int* __restrict__ token_list,
    float* __restrict__ weight_list, unsigned* __restrict__ xb, int T) {
  const int wave = threadIdx.x >> 5;
  const int lane = threadIdx.x & 31;
  const int t = blockIdx.x * (blockDim.x >> 5) + wave;
  if (t >= T) return;

  float acc[E_EXPERTS];
#pragma unroll
  for (int e = 0; e < E_EXPERTS; ++e) acc[e] = 0.f;

  const float* xr = x + (size_t)t * D_DIM;
  for (int d = lane; d < D_DIM; d += 32) {
    float xv = xr[d];
#pragma unroll
    for (int e = 0; e < E_EXPERTS; ++e)
      acc[e] = __builtin_fmaf(xv, gate_w[e * D_DIM + d], acc[e]);
  }
  if (xb) {  // bf16 copy of this token's row (pairs packed into dwords)
    const float2* xr2 = (const float2*)xr;
    unsigned* xo = xb + (size_t)t * (D_DIM / 2);
#pragma unroll
    for (int d2 = lane, it = 0; it < (D_DIM / 2) / 32; ++it, d2 += 32) {
      float2 v = xr2[d2];
      xo[d2] = pack2(v.x, v.y);
    }
  }
#pragma unroll
  for (int e = 0; e < E_EXPERTS; ++e) {
#pragma unroll
    for (int off = 16; off > 0; off >>= 1)
      acc[e] += __shfl_xor(acc[e], off, 32);
    acc[e] += gate_b[e];
  }

  if (lane == 0) {
#pragma unroll
    for (int e = 0; e < E_EXPERTS; ++e)
      logits_out[(size_t)t * E_EXPERTS + e] = acc[e];
    float b0 = -3.0e38f, b1 = -3.0e38f;
    int i0 = 0, i1 = 0;
#pragma unroll
    for (int e = 0; e < E_EXPERTS; ++e) {
      float v = acc[e];
      if (v > b0) { b1 = b0; i1 = i0; b0 = v; i0 = e; }
      else if (v > b1) { b1 = v; i1 = e; }
    }
    float w0 = 1.f / (1.f + __expf(b1 - b0));  // renormalized top-2 softmax
    float w1 = 1.f - w0;
    int s0 = atomicAdd(&counts[i0], 1);
    token_list[i0 * T + s0] = t; weight_list[i0 * T + s0] = w0;
    int s1 = atomicAdd(&counts[i1], 1);
    token_list[i1 * T + s1] = t; weight_list[i1 * T + s1] = w1;
  }
}

// ---------------------------------------------------------------------------
// expert_w fp32 -> packed bf16 pairs
// ---------------------------------------------------------------------------
__global__ __launch_bounds__(256) void convert_wb(const float* __restrict__ w,
                                                  unsigned* __restrict__ wb,
                                                  long long npairs) {
  long long i = (long long)blockIdx.x * blockDim.x + threadIdx.x;
  long long stride = (long long)gridDim.x * blockDim.x;
  const float2* w2 = (const float2*)w;
  for (; i < npairs; i += stride) {
    float2 v = w2[i];
    wb[i] = pack2(v.x, v.y);
  }
}

// ---------------------------------------------------------------------------
// Kernel 2 (fast path): gathered expert GEMM, bf16 WMMA, TDM-staged tiles,
// double-buffered. Block = 256 thr (8 waves), tile 128x128, K-step 32.
// Wave w -> rows (w&3)*32..+32 , cols (w>>2)*64..+64 : 8 WMMAs / K-step.
// A tile: TDM gather mode (16 token indices per wave descriptor).
// B tile: TDM 2D strided tile (16 rows per wave descriptor).
// ---------------------------------------------------------------------------
__global__ __launch_bounds__(256) void moe_expert_gemm_tdm(
    const unsigned* __restrict__ xb, const unsigned* __restrict__ wb,
    const float* __restrict__ expert_b, const int* __restrict__ counts,
    const int* __restrict__ token_list, const float* __restrict__ weight_list,
    float* __restrict__ out, int T, int O) {
  const int e = blockIdx.z;
  const int cnt = counts[e];
  const int m0 = blockIdx.x * TILE_M;
  if (m0 >= cnt) return;
  const int n0 = blockIdx.y * TILE_N;

  __shared__ unsigned At[2][TILE_M * RSTRIDE];
  __shared__ unsigned Bt[2][TILE_N * RSTRIDE];
  __shared__ int   tks[TILE_M];
  __shared__ float wts[TILE_M];

  const int tid = threadIdx.x;
  if (tid < TILE_M) {
    int row = m0 + tid;
    bool v = row < cnt;
    tks[tid] = v ? token_list[e * T + row] : 0;
    wts[tid] = v ? weight_list[e * T + row] : 0.f;
  }
  __syncthreads();

  const int wave = tid >> 5, lane = tid & 31;
  const int m = lane & 15, h = lane >> 4;
  const int mw = wave & 3, nw = wave >> 2;

  // Per-wave, K-invariant descriptor pieces -----------------------------
  unsigned idx[8];  // 16 x 16-bit gathered token indices for rows wave*16..+15
#pragma unroll
  for (int i = 0; i < 8; ++i) {
    unsigned t0 = (unsigned)tks[wave * 16 + 2 * i];
    unsigned t1 = (unsigned)tks[wave * 16 + 2 * i + 1];
    idx[i] = (t0 & 0xffffu) | (t1 << 16);
  }
  const unsigned ldsA0 = lds_off(&At[0][0]) + wave * 16 * RSTRIDE * 4;
  const unsigned ldsA1 = lds_off(&At[1][0]) + wave * 16 * RSTRIDE * 4;
  const unsigned ldsB0 = lds_off(&Bt[0][0]) + wave * 16 * RSTRIDE * 4;
  const unsigned ldsB1 = lds_off(&Bt[1][0]) + wave * 16 * RSTRIDE * 4;
  const unsigned long long gaA0 = (unsigned long long)(size_t)xb;  // + k0*2 each step
  const unsigned long long gaB0 =
      (unsigned long long)(size_t)(wb + ((size_t)e * O + n0 + wave * 16) * (D_DIM / 2));
  const unsigned GATHER = 1u << 31;  // 16-bit indices

  // prologue: stage k0 = 0 into buffer 0
  tdm_issue(GATHER, ldsA0, gaA0, TILE_K, 16, D_DIM, idx);
  tdm_issue(0u, ldsB0, gaB0, TILE_K, 16, D_DIM, nullptr);

  v8f c[2][4] = {};
  int buf = 0;
  for (int k0 = 0; k0 < D_DIM; k0 += TILE_K) {
    if (k0 + TILE_K < D_DIM) {  // prefetch next K-step into other buffer
      unsigned long long ka = (unsigned long long)(k0 + TILE_K) * 2ull;
      tdm_issue(GATHER, buf ? ldsA0 : ldsA1, gaA0 + ka, TILE_K, 16, D_DIM, idx);
      tdm_issue(0u, buf ? ldsB0 : ldsB1, gaB0 + ka, TILE_K, 16, D_DIM, nullptr);
      __builtin_amdgcn_s_wait_tensorcnt(2);  // current buffer's pair done
    } else {
      __builtin_amdgcn_s_wait_tensorcnt(0);
    }
    __syncthreads();

    const unsigned* A = &At[buf][0];
    const unsigned* B = &Bt[buf][0];
    // A frags (16-bit 16x32 layout): vj pair p = (j<4 ? j+4h : 4+j+4h)
    FragU a0, a1;
#pragma unroll
    for (int j = 0; j < 8; ++j) {
      int p = (j < 4) ? (j + 4 * h) : (4 + j + 4 * h);
      a0.u[j] = A[(mw * 32 + m) * RSTRIDE + p];
      a1.u[j] = A[(mw * 32 + 16 + m) * RSTRIDE + p];
    }
#pragma unroll
    for (int s = 0; s < 4; ++s) {
      // B frag (32x16 layout): vj holds K pair (j<4 ? j+4h : 4+j+4h) of output column
      FragU b;
#pragma unroll
      for (int j = 0; j < 8; ++j)
        b.u[j] = B[(nw * 64 + s * 16 + m) * RSTRIDE + ((j < 4) ? (j + 4 * h) : (4 + j + 4 * h))];
      c[0][s] = __builtin_amdgcn_wmma_f32_16x16x32_bf16(
          false, a0.h, false, b.h, (short)0, c[0][s], false, false);
      c[1][s] = __builtin_amdgcn_wmma_f32_16x16x32_bf16(
          false, a1.h, false, b.h, (short)0, c[1][s], false, false);
    }
    __syncthreads();
    buf ^= 1;
  }

  // Epilogue: C layout lane-half h, VGPR r -> M = r + 8h, N = lane&15.
#pragma unroll
  for (int a = 0; a < 2; ++a) {
    int rowv[8]; float roww[8]; bool rowok[8];
#pragma unroll
    for (int r = 0; r < 8; ++r) {
      int row = mw * 32 + a * 16 + r + 8 * h;
      rowok[r] = (m0 + row) < cnt;
      rowv[r] = tks[row];
      roww[r] = wts[row];
    }
#pragma unroll
    for (int s = 0; s < 4; ++s) {
      int col = n0 + nw * 64 + s * 16 + m;
      float bias = expert_b[e * O + col];
#pragma unroll
      for (int r = 0; r < 8; ++r)
        if (rowok[r])
          unsafeAtomicAdd(out + (size_t)rowv[r] * O + col,
                          roww[r] * (c[a][s][r] + bias));
    }
  }
}

// ---------------------------------------------------------------------------
// Fallback GEMM (fp32 sources, in-loop bf16 convert) if ws is too small.
// ---------------------------------------------------------------------------
__global__ __launch_bounds__(256) void moe_expert_gemm_f32(
    const float* __restrict__ x, const float* __restrict__ expert_w,
    const float* __restrict__ expert_b, const int* __restrict__ counts,
    const int* __restrict__ token_list, const float* __restrict__ weight_list,
    float* __restrict__ out, int T, int O) {
  const int e = blockIdx.z;
  const int cnt = counts[e];
  const int m0 = blockIdx.x * TILE_M;
  if (m0 >= cnt) return;
  const int n0 = blockIdx.y * 64;

  __shared__ unsigned Apk[TILE_M * RSTRIDE];
  __shared__ unsigned Bpk[16 * 65];
  __shared__ int   tks[TILE_M];
  __shared__ float wts[TILE_M];

  const int tid = threadIdx.x;
  if (tid < TILE_M) {
    int row = m0 + tid;
    bool v = row < cnt;
    tks[tid] = v ? token_list[e * T + row] : 0;
    wts[tid] = v ? weight_list[e * T + row] : 0.f;
  }
  __syncthreads();

  const int wave = tid >> 5, lane = tid & 31;
  const int m = lane & 15, h = lane >> 4;

  // hoist gather addresses out of the K-loop
  const float2* arow[8];
  unsigned avalid[8];
#pragma unroll
  for (int it = 0; it < 8; ++it) {
    int i = tid + it * 256;
    int row = i >> 4;
    arow[it] = (const float2*)(x + (size_t)tks[row] * D_DIM) + (i & 15);
    avalid[it] = ((m0 + row) < cnt) ? 0xffffffffu : 0u;
  }

  v8f c[4] = {};
  for (int k0 = 0; k0 < D_DIM; k0 += TILE_K) {
#pragma unroll
    for (int it = 0; it < 8; ++it) {
      int i = tid + it * 256;
      const float2 v = arow[it][k0 / 2];
      Apk[(i >> 4) * RSTRIDE + (i & 15)] = pack2(v.x, v.y) & avalid[it];
    }
#pragma unroll
    for (int it = 0; it < 4; ++it) {
      int i = tid + it * 256;
      int n = i >> 4, p = i & 15;
      const float2 v = *(const float2*)(expert_w +
                                        ((size_t)e * O + n0 + n) * D_DIM + k0 + 2 * p);
      Bpk[p * 65 + n] = pack2(v.x, v.y);
    }
    __syncthreads();
    FragU a;
#pragma unroll
    for (int j = 0; j < 8; ++j) {
      int p = (j < 4) ? (j + 4 * h) : (4 + j + 4 * h);
      a.u[j] = Apk[(wave * 16 + m) * RSTRIDE + p];
    }
#pragma unroll
    for (int s = 0; s < 4; ++s) {
      FragU b;
#pragma unroll
      for (int j = 0; j < 8; ++j) b.u[j] = Bpk[((j < 4) ? (j + 4 * h) : (4 + j + 4 * h)) * 65 + s * 16 + m];
      c[s] = __builtin_amdgcn_wmma_f32_16x16x32_bf16(
          false, a.h, false, b.h, (short)0, c[s], false, false);
    }
    __syncthreads();
  }
  int rowv[8]; float roww[8]; bool rowok[8];
#pragma unroll
  for (int r = 0; r < 8; ++r) {
    int row = wave * 16 + r + 8 * h;
    rowok[r] = (m0 + row) < cnt;
    rowv[r] = tks[row];
    roww[r] = wts[row];
  }
#pragma unroll
  for (int s = 0; s < 4; ++s) {
    int col = n0 + s * 16 + m;
    float bias = expert_b[e * O + col];
#pragma unroll
    for (int r = 0; r < 8; ++r)
      if (rowok[r])
        unsafeAtomicAdd(out + (size_t)rowv[r] * O + col,
                        roww[r] * (c[s][r] + bias));
  }
}

// ---------------------------------------------------------------------------
extern "C" void kernel_launch(void* const* d_in, const int* in_sizes, int n_in,
                              void* d_out, int out_size, void* d_ws,
                              size_t ws_size, hipStream_t stream) {
  const float* x        = (const float*)d_in[0];
  const float* gate_w   = (const float*)d_in[1];
  const float* gate_b   = (const float*)d_in[2];
  const float* expert_w = (const float*)d_in[3];
  const float* expert_b = (const float*)d_in[4];

  const int D = in_sizes[1] / E_EXPERTS;        // 1024
  const int T = in_sizes[0] / D;                // 8192
  const int O = in_sizes[3] / (E_EXPERTS * D);  // 1024
  (void)n_in; (void)out_size;

  float* out    = (float*)d_out;
  float* logits = out + (size_t)T * O;

  char*  ws          = (char*)d_ws;
  int*   counts      = (int*)ws;
  int*   token_list  = (int*)(ws + 128);
  float* weight_list = (float*)(ws + 128 + sizeof(int) * (size_t)E_EXPERTS * T);
  size_t lists_end   = 128 + 2 * sizeof(int) * (size_t)E_EXPERTS * T;
  size_t xb_off      = (lists_end + 255) & ~(size_t)255;
  size_t xb_bytes    = (size_t)T * D * 2;
  size_t wb_off      = xb_off + xb_bytes;
  size_t wb_bytes    = (size_t)E_EXPERTS * O * D * 2;
  bool fast = (wb_off + wb_bytes) <= ws_size;
  unsigned* xb = fast ? (unsigned*)(ws + xb_off) : nullptr;
  unsigned* wb = fast ? (unsigned*)(ws + wb_off) : nullptr;

  (void)hipMemsetAsync(counts, 0, 128, stream);
  (void)hipMemsetAsync(out, 0, sizeof(float) * (size_t)T * O, stream);

  if (fast)
    convert_wb<<<dim3(2048), dim3(256), 0, stream>>>(
        expert_w, wb, (long long)E_EXPERTS * O * D / 2);

  moe_router<<<dim3((T + 7) / 8), dim3(256), 0, stream>>>(
      x, gate_w, gate_b, logits, counts, token_list, weight_list, xb, T);

  if (fast) {
    dim3 grid(T / TILE_M, O / TILE_N, E_EXPERTS);
    moe_expert_gemm_tdm<<<grid, dim3(256), 0, stream>>>(
        xb, wb, expert_b, counts, token_list, weight_list, out, T, O);
  } else {
    dim3 grid(T / TILE_M, O / 64, E_EXPERTS);
    moe_expert_gemm_f32<<<grid, dim3(256), 0, stream>>>(
        x, expert_w, expert_b, counts, token_list, weight_list, out, T, O);
  }
}
